// MPNNBlock_5016521802505
// MI455X (gfx1250) — matrix operand order repaired and checked by reading, not compiled, not run on hardware
//
#include <hip/hip_runtime.h>

typedef __attribute__((ext_vector_type(16))) _Float16 v16h;
typedef __attribute__((ext_vector_type(8)))  float    v8f;

// ---------------------------------------------------------------------------
// Kernel 1: pack edge_W (1024x32 f32) into f16 WMMA B-fragments in d_ws.
// Fragment (d, nt): 32 lanes x 16 halves.  Lane L, element el holds
//   Wmat[(d,k), h] = edge_W[(d*32+h)*32 + k]
// with h = (L%16) + nt*16 and k = (el&7) + (el>>3)*16 + (L<16 ? 0 : 8)
// (B 32x16 f16 layout: lanes 0-15 hold K=0..7,16..23, lanes 16-31 hold K=8..15,24..31,
//  two K per VGPR, matching CDNA5 ISA B-matrix striping).
// Total: 32 d-steps * 2 n-tiles * 512 halves = 64 KB.
// ---------------------------------------------------------------------------
__global__ void mpnn_prep_B(const float* __restrict__ eW, _Float16* __restrict__ wsB) {
  int j = blockIdx.x * blockDim.x + threadIdx.x;   // 32768 total
  if (j >= 32768) return;
  int el   = j & 15;
  int lane = (j >> 4) & 31;
  int nt   = (j >> 9) & 1;
  int d    = j >> 10;
  int k = (el & 7) + ((el >> 3) << 4) + ((lane < 16) ? 0 : 8);
  int h = (lane & 15) + nt * 16;
  wsB[j] = (_Float16)eW[(d * 32 + h) * 32 + k];
}

// ---------------------------------------------------------------------------
// Kernel 2: out = x @ node_W^T  (exact fp32; also initializes d_out)
// ---------------------------------------------------------------------------
__global__ __launch_bounds__(256)
void mpnn_node(const float* __restrict__ x, const float* __restrict__ nW,
               float* __restrict__ out, int Nn) {
  __shared__ float sWT[32 * 32];            // transposed: sWT[d*32+h] = nW[h*32+d]
  for (int i = threadIdx.x; i < 1024; i += 256) {
    int dd = i >> 5, hh = i & 31;
    sWT[i] = nW[hh * 32 + dd];
  }
  __syncthreads();
  int t = blockIdx.x * 256 + threadIdx.x;
  int n = t >> 5, h = t & 31;
  if (n >= Nn) return;
  const float* xr = x + n * 32;
  float acc = 0.f;
  #pragma unroll
  for (int d = 0; d < 32; ++d) acc = fmaf(xr[d], sWT[d * 32 + h], acc);
  out[n * 32 + h] = acc;
}

// ---------------------------------------------------------------------------
// Kernel 3: fused message + scatter.
// msgs = P @ Wmat with P[e, d*32+k] = x[src[e],d] * edge_attr[e,k].
// Per wave: 2 M-tiles of 16 edges; K-loop of 32 steps; A-fragment for step d
// is the cached edge_attr fragment scaled by the f16 scalar x_j[row, d].
// Accumulate f32 via v_wmma_f32_16x16x32_f16, scatter with f32 atomics.
// ---------------------------------------------------------------------------
__global__ __launch_bounds__(256)
void mpnn_edge(const float* __restrict__ x,
               const float* __restrict__ eattr,
               const int* __restrict__ srcI,
               const int* __restrict__ dstI,
               const _Float16* __restrict__ wsB,
               float* __restrict__ out, int E_) {
  const int lane  = threadIdx.x & 31;
  const int wave  = threadIdx.x >> 5;
  const int wbase = (blockIdx.x * 8 + wave) * 32;   // 32 edges per wave
  const int row   = lane & 15;                      // A-matrix M row (both lane halves)
  const int koff  = (lane < 16) ? 0 : 8;            // A 16x32 f16 K striping

  const v16h* __restrict__ wsBv = (const v16h*)wsB;

  v16h     attr[2];
  _Float16 xh[2][32];
  v8f      c[2][2];

  #pragma unroll
  for (int t = 0; t < 2; ++t) {
    c[t][0] = {}; c[t][1] = {};
    int e = wbase + 16 * t + row;
    if (e > E_ - 1) e = E_ - 1;                      // clamp: keep EXEC all-1 for WMMA

    // edge_attr fragment: A layout -> lane needs k in {koff..koff+7, 16+koff..16+koff+7}
    const float* ap = eattr + e * 32 + koff;
    float4 r0 = *(const float4*)(ap + 0);
    float4 r1 = *(const float4*)(ap + 4);
    float4 r2 = *(const float4*)(ap + 16);
    float4 r3 = *(const float4*)(ap + 20);
    v16h a;
    a[0]=(_Float16)r0.x;  a[1]=(_Float16)r0.y;  a[2]=(_Float16)r0.z;  a[3]=(_Float16)r0.w;
    a[4]=(_Float16)r1.x;  a[5]=(_Float16)r1.y;  a[6]=(_Float16)r1.z;  a[7]=(_Float16)r1.w;
    a[8]=(_Float16)r2.x;  a[9]=(_Float16)r2.y;  a[10]=(_Float16)r2.z; a[11]=(_Float16)r2.w;
    a[12]=(_Float16)r3.x; a[13]=(_Float16)r3.y; a[14]=(_Float16)r3.z; a[15]=(_Float16)r3.w;
    attr[t] = a;

    // gathered x row (32 f32 -> f16 scalars, register-resident)
    int s = srcI[e];
    const float4* xp = (const float4*)(x + s * 32);
    #pragma unroll
    for (int q = 0; q < 8; ++q) {
      float4 f = xp[q];
      xh[t][4*q+0] = (_Float16)f.x;
      xh[t][4*q+1] = (_Float16)f.y;
      xh[t][4*q+2] = (_Float16)f.z;
      xh[t][4*q+3] = (_Float16)f.w;
    }
  }

  // K loop: 32 steps x (2 B-frag loads + 2 tiles x (8 pk_mul + 2 wmma))
  #pragma unroll
  for (int d = 0; d < 32; ++d) {
    v16h b0 = wsBv[(d * 2 + 0) * 32 + lane];
    v16h b1 = wsBv[(d * 2 + 1) * 32 + lane];
    #pragma unroll
    for (int t = 0; t < 2; ++t) {
      v16h a = attr[t] * xh[t][d];   // v_pk_mul_f16 x8 (scalar splat)
      c[t][0] = __builtin_amdgcn_wmma_f32_16x16x32_f16(
          false, a, false, b0, (short)0, c[t][0], false, false);
      c[t][1] = __builtin_amdgcn_wmma_f32_16x16x32_f16(
          false, a, false, b1, (short)0, c[t][1], false, false);
    }
  }

  // Scatter-add: C/D layout -> VGPR i, lanes 0-15: M=i; lanes 16-31: M=i+8; N=lane%16
  const int rowoff = (lane < 16) ? 0 : 8;
  const int h0 = lane & 15;
  #pragma unroll
  for (int t = 0; t < 2; ++t) {
    int ebase = wbase + 16 * t + rowoff;
    #pragma unroll
    for (int i = 0; i < 8; ++i) {
      int e = ebase + i;
      if (e < E_) {
        int dn = dstI[e];
        float* o = out + dn * 32 + h0;
        __hip_atomic_fetch_add(o,      c[t][0][i], __ATOMIC_RELAXED, __HIP_MEMORY_SCOPE_AGENT);
        __hip_atomic_fetch_add(o + 16, c[t][1][i], __ATOMIC_RELAXED, __HIP_MEMORY_SCOPE_AGENT);
      }
    }
  }
}

extern "C" void kernel_launch(void* const* d_in, const int* in_sizes, int n_in,
                              void* d_out, int out_size, void* d_ws, size_t ws_size,
                              hipStream_t stream) {
  const float* x     = (const float*)d_in[0];   // [N,32]
  const float* eattr = (const float*)d_in[1];   // [E,32]
  const float* eW    = (const float*)d_in[2];   // [1024,32]
  const float* nW    = (const float*)d_in[3];   // [32,32]
  const int*   eidx  = (const int*)d_in[4];     // [2,E]
  float* out = (float*)d_out;

  int Nn = in_sizes[0] / 32;
  int Ee = in_sizes[1] / 32;
  const int* src = eidx;
  const int* dst = eidx + Ee;
  _Float16* wsB = (_Float16*)d_ws;              // 64 KB of B fragments

  mpnn_prep_B<<<128, 256, 0, stream>>>(eW, wsB);
  mpnn_node<<<(Nn * 32 + 255) / 256, 256, 0, stream>>>(x, nW, out, Nn);
  mpnn_edge<<<(Ee + 255) / 256, 256, 0, stream>>>(x, eattr, src, dst, wsB, out, Ee);
}
